// CoarseAlignment_54331336294402
// MI455X (gfx1250) — matrix operand order repaired and checked by reading, not compiled, not run on hardware
//
#include <hip/hip_runtime.h>

typedef __attribute__((ext_vector_type(16))) _Float16 v16h;
typedef __attribute__((ext_vector_type(8)))  _Float16 v8h;
typedef __attribute__((ext_vector_type(8)))  float    v8f;

namespace {
constexpr int   kNSrc  = 8192;
constexpr int   kNDst  = 2048;
constexpr int   kD     = 1024;
constexpr int   kHImg  = 480;
constexpr int   kWImg  = 640;
constexpr int   kIter  = 10000;
constexpr float kInlTh = 0.05f;
constexpr float kEps   = 1e-8f;

// workspace layout (bytes)
constexpr size_t WS_FSRC  = 0;                                   // 8192*1024 f16
constexpr size_t WS_FDST  = WS_FSRC  + (size_t)kNSrc * kD * 2;   // 2048*1024 f16
constexpr size_t WS_ROWB  = WS_FDST  + (size_t)kNDst * kD * 2;   // 8192 u64
constexpr size_t WS_COLB  = WS_ROWB  + (size_t)kNSrc * 8;        // 2048 u64 (contiguous with ROWB)
constexpr size_t WS_NN12  = WS_COLB  + (size_t)kNDst * 8;        // 8192 i32
constexpr size_t WS_NN21  = WS_NN12  + (size_t)kNSrc * 4;        // 2048 i32
constexpr size_t WS_MUT   = WS_NN21  + (size_t)kNDst * 4;        // 8192 i32
constexpr size_t WS_HS    = WS_MUT   + (size_t)kNSrc * 4;        // 10000*9 f32
constexpr size_t WS_OK    = WS_HS    + (size_t)kIter * 9 * 4;    // 10000 i32
constexpr size_t WS_CNT   = WS_OK    + (size_t)kIter * 4;        // 10000 i32
constexpr size_t WS_BESTH = WS_CNT   + (size_t)kIter * 4;        // 9 f32 (+pad)
constexpr size_t WS_PART  = WS_BESTH + 64;                       // 32*45 f32
constexpr size_t WS_HFIN  = WS_PART  + (size_t)32 * 45 * 4;      // 9 f32
} // namespace

// ---------------- helpers ----------------

__device__ __forceinline__ unsigned long long packKey(float v, unsigned idx) {
  unsigned u = __float_as_uint(v);
  unsigned key = (u & 0x80000000u) ? ~u : (u | 0x80000000u); // monotone float order
  return ((unsigned long long)key << 32) | (unsigned long long)(0xFFFFFFFFu - idx);
}

// cyclic Jacobi on 9x9 symmetric; returns eigenvector of smallest eigenvalue
__device__ void eig9_smallest(float M[9][9], float h[9]) {
  float V[9][9];
  for (int i = 0; i < 9; ++i)
    for (int j = 0; j < 9; ++j) V[i][j] = (i == j) ? 1.f : 0.f;
  for (int sweep = 0; sweep < 14; ++sweep) {
    for (int p = 0; p < 8; ++p) {
      for (int q = p + 1; q < 9; ++q) {
        float apq = M[p][q];
        if (fabsf(apq) < 1e-20f) continue;
        float app = M[p][p], aqq = M[q][q];
        float tau = (aqq - app) / (2.f * apq);
        float t = (tau >= 0.f) ? 1.f / (tau + sqrtf(1.f + tau * tau))
                               : 1.f / (tau - sqrtf(1.f + tau * tau));
        float c = 1.f / sqrtf(1.f + t * t);
        float s = t * c;
        for (int i = 0; i < 9; ++i) {
          float mip = M[i][p], miq = M[i][q];
          M[i][p] = c * mip - s * miq;
          M[i][q] = s * mip + c * miq;
        }
        for (int i = 0; i < 9; ++i) {
          float mpi = M[p][i], mqi = M[q][i];
          M[p][i] = c * mpi - s * mqi;
          M[q][i] = s * mpi + c * mqi;
        }
        for (int i = 0; i < 9; ++i) {
          float vip = V[i][p], viq = V[i][q];
          V[i][p] = c * vip - s * viq;
          V[i][q] = s * vip + c * viq;
        }
      }
    }
  }
  int best = 0;
  float bv = M[0][0];
  for (int i = 1; i < 9; ++i)
    if (M[i][i] < bv) { bv = M[i][i]; best = i; }
  for (int i = 0; i < 9; ++i) h[i] = V[i][best];
}

__device__ __forceinline__ void dlt_rows(float u, float v, float x, float y,
                                         float r1[9], float r2[9]) {
  r1[0] = -u; r1[1] = -v; r1[2] = -1.f; r1[3] = 0.f; r1[4] = 0.f; r1[5] = 0.f;
  r1[6] = x * u; r1[7] = x * v; r1[8] = x;
  r2[0] = 0.f; r2[1] = 0.f; r2[2] = 0.f; r2[3] = -u; r2[4] = -v; r2[5] = -1.f;
  r2[6] = y * u; r2[7] = y * v; r2[8] = y;
}

// ---------------- kernels ----------------

// L2-normalize each row, convert to f16. One wave per row.
__global__ void k_normalize(const float* __restrict__ in, _Float16* __restrict__ out, int rows) {
  int gid = blockIdx.x * blockDim.x + threadIdx.x;
  int row = gid >> 5;
  int lane = gid & 31;
  if (row >= rows) return;
  const float* r = in + (size_t)row * kD;
  float ss = 0.f;
  for (int j = lane; j < kD; j += 32) { float x = r[j]; ss += x * x; }
  #pragma unroll
  for (int off = 16; off >= 1; off >>= 1) ss += __shfl_xor(ss, off);
  float inv = 1.f / (sqrtf(ss) + kEps);
  _Float16* o = out + (size_t)row * kD;
  for (int j = lane; j < kD; j += 32) o[j] = (_Float16)(r[j] * inv);
}

__global__ void k_init_best(unsigned long long* __restrict__ b, int n) {
  int i = blockIdx.x * blockDim.x + threadIdx.x;
  if (i < n) b[i] = 0ull;
}

// Reduce one 16x16 WMMA accumulator tile into row/col argmax accumulators.
// D layout: element (m,n) -> n = lane&15, m = vgpr + 8*(lane>>4)
__device__ __forceinline__ void tile_reduce(v8f acc, int tmTile, int tnTile, int lane,
                                            unsigned long long* __restrict__ row_best,
                                            unsigned long long* __restrict__ col_best) {
  const int half = lane >> 4;
  const int ln = lane & 15;

  // per-column (n) argmax over the tile's 16 rows
  float cmax = acc[0];
  int cargm = 8 * half;
  #pragma unroll
  for (int r = 1; r < 8; ++r) {
    float v = acc[r];
    if (v > cmax) { cmax = v; cargm = r + 8 * half; }
  }
  {
    float ov = __shfl_xor(cmax, 16);
    int   oa = __shfl_xor(cargm, 16);
    if (ov > cmax || (ov == cmax && oa < cargm)) { cmax = ov; cargm = oa; }
  }
  if (half == 0) {
    unsigned grow = (unsigned)(tmTile * 16 + cargm);
    atomicMax(col_best + tnTile * 16 + ln, packKey(cmax, grow));
  }

  // per-row (m) argmax over the tile's 16 cols (reduce within each 16-lane half)
  #pragma unroll
  for (int r = 0; r < 8; ++r) {
    float v = acc[r];
    int a = ln;
    #pragma unroll
    for (int off = 8; off >= 1; off >>= 1) {
      float sv = __shfl_xor(v, off);
      int   sa = __shfl_xor(a, off);
      if (sv > v || (sv == v && sa < a)) { v = sv; a = sa; }
    }
    if (ln == 0) {
      int m = r + 8 * half;
      atomicMax(row_best + tmTile * 16 + m, packKey(v, (unsigned)(tnTile * 16 + a)));
    }
  }
}

// Fused f16 WMMA GEMM (sim = A * B^T) + per-row / per-col argmax via packed atomicMax.
// One wave computes a 32x32 block (2x2 tiles, 4 accumulators) reusing A/B fragments:
// 4 WMMA per k-step against 4 b128-load-pairs -> 2x less L2 traffic per FLOP.
__global__ __launch_bounds__(128) void k_sim_wmma(
    const _Float16* __restrict__ A,   // [8192,1024] normalized src
    const _Float16* __restrict__ B,   // [2048,1024] normalized dst
    unsigned long long* __restrict__ row_best,
    unsigned long long* __restrict__ col_best) {
  const int lane = threadIdx.x & 31;
  const int wave = threadIdx.x >> 5;
  const int blk = blockIdx.x * 4 + wave;
  const int tn2 = blk & (kNDst / 32 - 1);   // 64 col blocks of 32
  const int tm2 = blk >> 6;                 // 256 row blocks of 32
  const int half = lane >> 4;
  const int ln = lane & 15;

  const _Float16* ap0 = A + (size_t)(tm2 * 32 + ln) * kD;
  const _Float16* ap1 = A + (size_t)(tm2 * 32 + 16 + ln) * kD;
  const _Float16* bp0 = B + (size_t)(tn2 * 32 + ln) * kD;
  const _Float16* bp1 = B + (size_t)(tn2 * 32 + 16 + ln) * kD;

  v8f acc00 = {}, acc01 = {}, acc10 = {}, acc11 = {};
  for (int k = 0; k < kD; k += 32) {
    // A fragment (16x32 f16): lane row = ln; half selects K 0-15 / 16-31 (contiguous)
    v16h af0 = *(const v16h*)(ap0 + k + half * 16);
    v16h af1 = *(const v16h*)(ap1 + k + half * 16);
    // B fragment (32x16 f16): lane col = ln; half selects K 0-15 / 16-31 (contiguous)
    v16h bf0 = *(const v16h*)(bp0 + k + half * 16);
    v16h bf1 = *(const v16h*)(bp1 + k + half * 16);

    acc00 = __builtin_amdgcn_wmma_f32_16x16x32_f16(false, af0, false, bf0,
                                                   (short)0, acc00, false, false);
    acc01 = __builtin_amdgcn_wmma_f32_16x16x32_f16(false, af0, false, bf1,
                                                   (short)0, acc01, false, false);
    acc10 = __builtin_amdgcn_wmma_f32_16x16x32_f16(false, af1, false, bf0,
                                                   (short)0, acc10, false, false);
    acc11 = __builtin_amdgcn_wmma_f32_16x16x32_f16(false, af1, false, bf1,
                                                   (short)0, acc11, false, false);

    if (k + 64 < kD) {
      __builtin_prefetch(ap0 + k + 64, 0, 1);
      __builtin_prefetch(ap1 + k + 64, 0, 1);
      __builtin_prefetch(bp0 + k + 64, 0, 1);
      __builtin_prefetch(bp1 + k + 64, 0, 1);
    }
  }

  tile_reduce(acc00, tm2 * 2 + 0, tn2 * 2 + 0, lane, row_best, col_best);
  tile_reduce(acc01, tm2 * 2 + 0, tn2 * 2 + 1, lane, row_best, col_best);
  tile_reduce(acc10, tm2 * 2 + 1, tn2 * 2 + 0, lane, row_best, col_best);
  tile_reduce(acc11, tm2 * 2 + 1, tn2 * 2 + 1, lane, row_best, col_best);
}

__global__ void k_extract_nn(const unsigned long long* __restrict__ rb,
                             const unsigned long long* __restrict__ cb,
                             int* __restrict__ nn12, int* __restrict__ nn21) {
  int i = blockIdx.x * blockDim.x + threadIdx.x;
  if (i < kNSrc) nn12[i] = (int)(0xFFFFFFFFu - (unsigned)(rb[i] & 0xFFFFFFFFull));
  if (i < kNDst) nn21[i] = (int)(0xFFFFFFFFu - (unsigned)(cb[i] & 0xFFFFFFFFull));
}

__global__ void k_mutual(const int* __restrict__ nn12, const int* __restrict__ nn21,
                         int* __restrict__ mutual) {
  int i = blockIdx.x * blockDim.x + threadIdx.x;
  if (i < kNSrc) mutual[i] = (nn21[nn12[i]] == i) ? 1 : 0;
}

// one thread per RANSAC hypothesis: 4-point DLT via A^T A + Jacobi
__global__ void k_dlt_samples(const float* __restrict__ h_src, const float* __restrict__ w_src,
                              const float* __restrict__ h_dst, const float* __restrict__ w_dst,
                              const int* __restrict__ smp, const int* __restrict__ nn12,
                              const int* __restrict__ mutual,
                              float* __restrict__ Hs, int* __restrict__ sample_ok) {
  int c = blockIdx.x * blockDim.x + threadIdx.x;
  if (c >= kIter) return;
  float M[9][9];
  for (int i = 0; i < 9; ++i)
    for (int j = 0; j < 9; ++j) M[i][j] = 0.f;
  int ok = 1;
  for (int j = 0; j < 4; ++j) {
    int id = smp[c * 4 + j];
    ok &= mutual[id];
    float u = h_src[id], v = w_src[id];
    int d = nn12[id];
    float x = h_dst[d], y = w_dst[d];
    float r1[9], r2[9];
    dlt_rows(u, v, x, y, r1, r2);
    for (int a = 0; a < 9; ++a)
      for (int b = 0; b < 9; ++b) M[a][b] += r1[a] * r1[b] + r2[a] * r2[b];
  }
  float h[9];
  eig9_smallest(M, h);
  float dn = h[8] + kEps;
  for (int t = 0; t < 9; ++t) Hs[(size_t)c * 9 + t] = h[t] / dn;
  sample_ok[c] = ok;
}

// one block per hypothesis: count inliers over all 8192 points
__global__ __launch_bounds__(256) void k_count_inliers(
    const float* __restrict__ Hs, const int* __restrict__ sample_ok,
    const float* __restrict__ h_src, const float* __restrict__ w_src,
    const float* __restrict__ h_dst, const float* __restrict__ w_dst,
    const int* __restrict__ nn12, const int* __restrict__ mutual,
    int* __restrict__ counts) {
  int c = blockIdx.x;
  int tid = threadIdx.x;
  float H0 = Hs[c * 9 + 0], H1 = Hs[c * 9 + 1], H2 = Hs[c * 9 + 2];
  float H3 = Hs[c * 9 + 3], H4 = Hs[c * 9 + 4], H5 = Hs[c * 9 + 5];
  float H6 = Hs[c * 9 + 6], H7 = Hs[c * 9 + 7], H8 = Hs[c * 9 + 8];
  int cnt = 0;
  for (int i = tid; i < kNSrc; i += 256) {
    float u = h_src[i], v = w_src[i];
    float t2 = H6 * u + H7 * v + H8;
    float inv = 1.f / (t2 + kEps);
    float pr0 = (H0 * u + H1 * v + H2) * inv;
    float pr1 = (H3 * u + H4 * v + H5) * inv;
    int d = nn12[i];
    float dx = pr0 - h_dst[d];
    float dy = pr1 - w_dst[d];
    float err = dx * dx + dy * dy;
    cnt += ((err < kInlTh) && mutual[i]) ? 1 : 0;
  }
  __shared__ int red[256];
  red[tid] = cnt;
  __syncthreads();
  for (int s = 128; s >= 1; s >>= 1) {
    if (tid < s) red[tid] += red[tid + s];
    __syncthreads();
  }
  if (tid == 0) counts[c] = sample_ok[c] ? red[0] : -1;
}

// single block: argmax over counts (ties -> first), copy best H
__global__ __launch_bounds__(256) void k_argmax_counts(
    const int* __restrict__ counts, const float* __restrict__ Hs,
    float* __restrict__ bestH) {
  int tid = threadIdx.x;
  int best = -2, bidx = 0;
  for (int i = tid; i < kIter; i += 256) {
    int v = counts[i];
    if (v > best || (v == best && i < bidx)) { best = v; bidx = i; }
  }
  __shared__ int sc[256];
  __shared__ int si[256];
  sc[tid] = best; si[tid] = bidx;
  __syncthreads();
  for (int s = 128; s >= 1; s >>= 1) {
    if (tid < s) {
      if (sc[tid + s] > sc[tid] || (sc[tid + s] == sc[tid] && si[tid + s] < si[tid])) {
        sc[tid] = sc[tid + s]; si[tid] = si[tid + s];
      }
    }
    __syncthreads();
  }
  if (tid < 9) bestH[tid] = Hs[(size_t)si[0] * 9 + tid];
}

// 32 blocks * 256 threads = 8192 points; deterministic partial A^T A (45 upper-tri)
__global__ __launch_bounds__(256) void k_ata_partials(
    const float* __restrict__ bestH,
    const float* __restrict__ h_src, const float* __restrict__ w_src,
    const float* __restrict__ h_dst, const float* __restrict__ w_dst,
    const int* __restrict__ nn12, const int* __restrict__ mutual,
    float* __restrict__ partials) {
  int tid = threadIdx.x;
  int i = blockIdx.x * 256 + tid;
  float H0 = bestH[0], H1 = bestH[1], H2 = bestH[2];
  float H3 = bestH[3], H4 = bestH[4], H5 = bestH[5];
  float H6 = bestH[6], H7 = bestH[7], H8 = bestH[8];
  float u = h_src[i], v = w_src[i];
  int d = nn12[i];
  float x = h_dst[d], y = w_dst[d];
  float t2 = H6 * u + H7 * v + H8;
  float inv = 1.f / (t2 + kEps);
  float pr0 = (H0 * u + H1 * v + H2) * inv;
  float pr1 = (H3 * u + H4 * v + H5) * inv;
  float dx = pr0 - x, dy = pr1 - y;
  float err = dx * dx + dy * dy;
  float w = ((err < kInlTh) && mutual[i]) ? 1.f : 0.f;
  float r1[9], r2[9];
  dlt_rows(u, v, x, y, r1, r2);
  float loc[45];
  {
    int t = 0;
    for (int a = 0; a < 9; ++a)
      for (int b = a; b < 9; ++b) loc[t++] = w * (r1[a] * r1[b] + r2[a] * r2[b]);
  }
  // wave-level sum of each component (deterministic shuffle tree)
  for (int j = 0; j < 45; ++j) {
    float s = loc[j];
    #pragma unroll
    for (int off = 16; off >= 1; off >>= 1) s += __shfl_xor(s, off);
    loc[j] = s;
  }
  __shared__ float lds[8 * 45];
  int lane = tid & 31, wid = tid >> 5;
  if (lane == 0)
    for (int j = 0; j < 45; ++j) lds[wid * 45 + j] = loc[j];
  __syncthreads();
  if (tid < 45) {
    float s = 0.f;
    for (int wv = 0; wv < 8; ++wv) s += lds[wv * 45 + tid];  // fixed order
    partials[blockIdx.x * 45 + tid] = s;
  }
}

// single block: sum 32 partials (fixed order), Jacobi, write H_final
__global__ __launch_bounds__(64) void k_final_dlt(const float* __restrict__ partials,
                                                  float* __restrict__ Hfin) {
  int tid = threadIdx.x;
  __shared__ float Ms[45];
  if (tid < 45) {
    float s = 0.f;
    for (int b = 0; b < 32; ++b) s += partials[b * 45 + tid];
    Ms[tid] = s;
  }
  __syncthreads();
  if (tid == 0) {
    float M[9][9];
    int t = 0;
    for (int a = 0; a < 9; ++a)
      for (int b = a; b < 9; ++b) { M[a][b] = Ms[t]; M[b][a] = Ms[t]; ++t; }
    float h[9];
    eig9_smallest(M, h);
    float dn = h[8] + kEps;
    for (int j = 0; j < 9; ++j) Hfin[j] = h[j] / dn;
  }
}

// warp + bilinear sample: one thread per output pixel
__global__ __launch_bounds__(256) void k_warp(const float* __restrict__ I,
                                              const float* __restrict__ Hf,
                                              float* __restrict__ out) {
  int idx = blockIdx.x * blockDim.x + threadIdx.x;
  if (idx >= kHImg * kWImg) return;
  int py = idx / kWImg, px = idx - py * kWImg;
  float gx = -1.f + 2.f * (float)px / (float)(kWImg - 1);
  float gy = -1.f + 2.f * (float)py / (float)(kHImg - 1);
  float H0 = Hf[0], H1 = Hf[1], H2 = Hf[2];
  float H3 = Hf[3], H4 = Hf[4], H5 = Hf[5];
  float H6 = Hf[6], H7 = Hf[7], H8 = Hf[8];
  float t0 = H0 * gx + H1 * gy + H2;
  float t1 = H3 * gx + H4 * gy + H5;
  float t2 = H6 * gx + H7 * gy + H8;
  float inv = 1.f / (t2 + kEps);
  float gx2 = t0 * inv, gy2 = t1 * inv;
  float x = (gx2 + 1.f) * 0.5f * (float)(kWImg - 1);
  float y = (gy2 + 1.f) * 0.5f * (float)(kHImg - 1);
  float x0 = floorf(x), y0 = floorf(y);
  float x1 = x0 + 1.f, y1 = y0 + 1.f;
  float wx1 = x - x0, wy1 = y - y0;
  float wx0 = 1.f - wx1, wy0 = 1.f - wy1;

  float r[3] = {0.f, 0.f, 0.f};
  #pragma unroll
  for (int corner = 0; corner < 4; ++corner) {
    float xi = (corner & 1) ? x1 : x0;
    float yi = (corner & 2) ? y1 : y0;
    float wgt = ((corner & 1) ? wx1 : wx0) * ((corner & 2) ? wy1 : wy0);
    bool inb = (xi >= 0.f) && (xi <= (float)(kWImg - 1)) &&
               (yi >= 0.f) && (yi <= (float)(kHImg - 1));
    int xc = (int)fminf(fmaxf(xi, 0.f), (float)(kWImg - 1));
    int yc = (int)fminf(fmaxf(yi, 0.f), (float)(kHImg - 1));
    float m = inb ? wgt : 0.f;
    size_t base = (size_t)yc * kWImg + xc;
    r[0] += I[0 * (size_t)kHImg * kWImg + base] * m;
    r[1] += I[1 * (size_t)kHImg * kWImg + base] * m;
    r[2] += I[2 * (size_t)kHImg * kWImg + base] * m;
  }
  out[0 * (size_t)kHImg * kWImg + idx] = r[0];
  out[1 * (size_t)kHImg * kWImg + idx] = r[1];
  out[2 * (size_t)kHImg * kWImg + idx] = r[2];
}

// ---------------- launcher ----------------

extern "C" void kernel_launch(void* const* d_in, const int* in_sizes, int n_in,
                              void* d_out, int out_size, void* d_ws, size_t ws_size,
                              hipStream_t stream) {
  (void)in_sizes; (void)n_in; (void)out_size; (void)ws_size;
  const float* I_src = (const float*)d_in[0];
  const float* fsrc  = (const float*)d_in[1];
  const float* fdst  = (const float*)d_in[2];
  const float* h_src = (const float*)d_in[3];
  const float* w_src = (const float*)d_in[4];
  const float* h_dst = (const float*)d_in[5];
  const float* w_dst = (const float*)d_in[6];
  const int*   smp   = (const int*)d_in[7];
  float* out = (float*)d_out;

  char* ws = (char*)d_ws;
  _Float16* fsrc_h = (_Float16*)(ws + WS_FSRC);
  _Float16* fdst_h = (_Float16*)(ws + WS_FDST);
  unsigned long long* row_best = (unsigned long long*)(ws + WS_ROWB);
  unsigned long long* col_best = (unsigned long long*)(ws + WS_COLB);
  int*   nn12      = (int*)(ws + WS_NN12);
  int*   nn21      = (int*)(ws + WS_NN21);
  int*   mutual    = (int*)(ws + WS_MUT);
  float* Hs        = (float*)(ws + WS_HS);
  int*   sample_ok = (int*)(ws + WS_OK);
  int*   counts    = (int*)(ws + WS_CNT);
  float* bestH     = (float*)(ws + WS_BESTH);
  float* partials  = (float*)(ws + WS_PART);
  float* Hfin      = (float*)(ws + WS_HFIN);

  // 1) normalize features -> f16 (one wave per row)
  k_normalize<<<kNSrc / 8, 256, 0, stream>>>(fsrc, fsrc_h, kNSrc);
  k_normalize<<<kNDst / 8, 256, 0, stream>>>(fdst, fdst_h, kNDst);

  // 2) init argmax accumulators (row_best + col_best contiguous)
  k_init_best<<<(kNSrc + kNDst + 255) / 256, 256, 0, stream>>>(row_best, kNSrc + kNDst);

  // 3) fused WMMA GEMM (32x32 per wave) + bidirectional argmax
  k_sim_wmma<<<(kNSrc / 32) * (kNDst / 32) / 4, 128, 0, stream>>>(fsrc_h, fdst_h,
                                                                  row_best, col_best);

  // 4) nearest neighbors + mutual check
  k_extract_nn<<<kNSrc / 256, 256, 0, stream>>>(row_best, col_best, nn12, nn21);
  k_mutual<<<kNSrc / 256, 256, 0, stream>>>(nn12, nn21, mutual);

  // 5) per-hypothesis 4-point DLT
  k_dlt_samples<<<(kIter + 255) / 256, 256, 0, stream>>>(h_src, w_src, h_dst, w_dst,
                                                         smp, nn12, mutual, Hs, sample_ok);

  // 6) inlier counting (one block per hypothesis)
  k_count_inliers<<<kIter, 256, 0, stream>>>(Hs, sample_ok, h_src, w_src, h_dst, w_dst,
                                             nn12, mutual, counts);

  // 7) select best hypothesis
  k_argmax_counts<<<1, 256, 0, stream>>>(counts, Hs, bestH);

  // 8) weighted A^T A over all points (deterministic two-pass reduction)
  k_ata_partials<<<32, 256, 0, stream>>>(bestH, h_src, w_src, h_dst, w_dst,
                                         nn12, mutual, partials);
  k_final_dlt<<<1, 64, 0, stream>>>(partials, Hfin);

  // 9) warp + bilinear sample
  k_warp<<<(kHImg * kWImg + 255) / 256, 256, 0, stream>>>(I_src, Hfin, out);
}
